// OuterProductMean_62663572848691
// MI455X (gfx1250) — matrix operand order repaired and checked by reading, not compiled, not run on hardware
//
#include <hip/hip_runtime.h>
#include <hip/hip_bf16.h>

typedef __attribute__((ext_vector_type(16))) _Float16 v16h;
typedef __attribute__((ext_vector_type(8)))  float    v8f;

#define NRR   512
#define NLL   256
#define MM    4
#define INC   256
#define MIDC  32
#define OUTC  128
#define NTOT  768   // NR + NL

union FragAB { v16h h; uint4 q[2]; };

// ---------------------------------------------------------------------------
// Kernel 1: layernorm + projection for receptor rows.
// One block per residue i (512 blocks), 128 threads (one per output channel).
// Channel t maps to (x = t>>2, slice = t&3): slices 0..3 = r_ri, r_rj, r_li, r_lj.
// ---------------------------------------------------------------------------
__global__ __launch_bounds__(128) void ln_proj_rec_kernel(
    const float* __restrict__ rec, const float* __restrict__ g, const float* __restrict__ b,
    const float* __restrict__ W,   const float* __restrict__ bias,
    _Float16* __restrict__ a_rr, float* __restrict__ r_rj,
    _Float16* __restrict__ a_rl, _Float16* __restrict__ a_lr)
{
    __shared__ float xs[INC];
    __shared__ float red[128];
    const int i = blockIdx.x;
    const int t = threadIdx.x;

    const float* x = rec + (size_t)i * INC;
    float v0 = x[t], v1 = x[t + 128];

    red[t] = v0 + v1;
    __syncthreads();
    for (int s = 64; s > 0; s >>= 1) { if (t < s) red[t] += red[t + s]; __syncthreads(); }
    float mu = red[0] * (1.0f / INC);
    __syncthreads();

    float c0 = v0 - mu, c1 = v1 - mu;
    red[t] = c0 * c0 + c1 * c1;
    __syncthreads();
    for (int s = 64; s > 0; s >>= 1) { if (t < s) red[t] += red[t + s]; __syncthreads(); }
    float rstd = rsqrtf(red[0] * (1.0f / INC) + 1e-5f);
    __syncthreads();

    xs[t]       = c0 * rstd * g[t]       + b[t];
    xs[t + 128] = c1 * rstd * g[t + 128] + b[t + 128];
    __syncthreads();

    const float* w = W + (size_t)t * INC;
    float acc = bias[t];
    #pragma unroll 4
    for (int c = 0; c < INC; ++c) acc += w[c] * xs[c];

    const int s = t & 3, xn = t >> 2;
    const int idx = i * MIDC + xn;
    if      (s == 0) a_rr[idx] = (_Float16)acc;   // r_ri
    else if (s == 1) r_rj[idx] = acc;             // r_rj (fp32 for t-compute)
    else if (s == 2) a_rl[idx] = (_Float16)acc;   // r_li
    else             a_lr[idx] = (_Float16)acc;   // r_lj
}

// ---------------------------------------------------------------------------
// Kernel 2: layernorm + projection for ligand rows, looped over m copies.
// One block per ligand residue i (256 blocks), 128 threads.
// ---------------------------------------------------------------------------
__global__ __launch_bounds__(128) void ln_proj_lig_kernel(
    const float* __restrict__ lig, const float* __restrict__ g, const float* __restrict__ b,
    const float* __restrict__ W,   const float* __restrict__ bias,
    float* __restrict__ l_ri_m, float* __restrict__ l_rj_m,
    _Float16* __restrict__ a_ll, float* __restrict__ l_lj)
{
    __shared__ float xs[INC];
    __shared__ float red[128];
    const int i = blockIdx.x;
    const int t = threadIdx.x;
    const int s = t & 3, xn = t >> 2;

    float accMean = 0.0f;

    for (int m = 0; m < MM; ++m) {
        const float* x = lig + ((size_t)(m * NLL) + i) * INC;
        float v0 = x[t], v1 = x[t + 128];

        red[t] = v0 + v1;
        __syncthreads();
        for (int ss = 64; ss > 0; ss >>= 1) { if (t < ss) red[t] += red[t + ss]; __syncthreads(); }
        float mu = red[0] * (1.0f / INC);
        __syncthreads();

        float c0 = v0 - mu, c1 = v1 - mu;
        red[t] = c0 * c0 + c1 * c1;
        __syncthreads();
        for (int ss = 64; ss > 0; ss >>= 1) { if (t < ss) red[t] += red[t + ss]; __syncthreads(); }
        float rstd = rsqrtf(red[0] * (1.0f / INC) + 1e-5f);
        __syncthreads();

        xs[t]       = c0 * rstd * g[t]       + b[t];
        xs[t + 128] = c1 * rstd * g[t + 128] + b[t + 128];
        __syncthreads();

        const float* w = W + (size_t)t * INC;
        float acc = bias[t];
        #pragma unroll 4
        for (int c = 0; c < INC; ++c) acc += w[c] * xs[c];

        if      (s == 0) accMean += acc;                                        // l_ri
        else if (s == 1) accMean += acc;                                        // l_rj
        else if (s == 2) a_ll[i * (MM * MIDC) + m * MIDC + xn] = (_Float16)acc; // l_li, K=m*32+x
        else             l_lj[((size_t)(m * NLL) + i) * MIDC + xn] = acc;       // l_lj
        __syncthreads();
    }

    if      (s == 0) l_ri_m[i * MIDC + xn] = accMean * (1.0f / MM);
    else if (s == 1) l_rj_m[i * MIDC + xn] = accMean * (1.0f / MM);
}

// ---------------------------------------------------------------------------
// Kernel 3: t[j][o][x] = scale * sum_y src[j][y] * W[o][x*32 + y], stored f16
// in layout dst[(j*128 + o)*KKdst + kOff + x] so WMMA B fragments are
// contiguous-per-lane loads. One block per j, 256 threads.
// ---------------------------------------------------------------------------
__global__ __launch_bounds__(256) void compute_t_kernel(
    const float* __restrict__ src, const float* __restrict__ W,
    _Float16* __restrict__ dst, int KKdst, int kOff, float scale)
{
    __shared__ float sv[MIDC];
    const int j = blockIdx.x;
    const int t = threadIdx.x;
    if (t < MIDC) sv[t] = src[j * MIDC + t];
    __syncthreads();

    for (int idx = t; idx < OUTC * MIDC; idx += 256) {
        const int o  = idx >> 5;
        const int xn = idx & 31;
        const float* w = W + (size_t)o * (MIDC * MIDC) + xn * MIDC;
        float acc = 0.0f;
        #pragma unroll
        for (int y = 0; y < MIDC; ++y) acc += w[y] * sv[y];
        dst[((size_t)(j * OUTC + o)) * KKdst + kOff + xn] = (_Float16)(acc * scale);
    }
}

// ---------------------------------------------------------------------------
// Kernel 4: the pairwise WMMA kernel.
//   out[row, col, o] = bias[o] + sum_k A[i][k] * BT[j][o][k]
// with (row,col) = (rowBase+i, colBase+j) or swapped (rowBase+j, colBase+i).
// Grid = J * (numI/128) blocks, 256 threads (8 waves); each wave owns a
// 16-row strip x all 128 output channels (8 WMMA tiles, K accumulated).
// Fragment loads follow the CDNA5 16-bit A / B VGPR layouts directly from
// global memory (each lane: two 16B contiguous chunks). Output stores are
// non-temporal: the 302 MB result is write-once, so keep it out of L2 and
// preserve residency for the heavily reused bt_* operand tables.
// ---------------------------------------------------------------------------
__global__ __launch_bounds__(256) void pair_wmma_kernel(
    const _Float16* __restrict__ A,   // [numI][KK] row-major f16
    const _Float16* __restrict__ BT,  // [J][128][KK] f16
    const float* __restrict__ bias,   // [128]
    float* __restrict__ out,          // [768][768][128]
    int numItiles, int J, int KK, int rowBase, int colBase, int swapped)
{
    const int j    = blockIdx.x / numItiles;
    const int it   = blockIdx.x % numItiles;
    const int wave = threadIdx.x >> 5;
    const int lane = threadIdx.x & 31;
    const int m    = lane & 15;   // row within A tile / column within B,D tile
    const int hi   = lane >> 4;   // K-half selector (A,B) / row-half (D)
    const int iw   = it * 128 + wave * 16;

    v8f acc[8];
    const v8f zero = {0.f, 0.f, 0.f, 0.f, 0.f, 0.f, 0.f, 0.f};
    #pragma unroll
    for (int ot = 0; ot < 8; ++ot) acc[ot] = zero;

    const _Float16* arow = A  + (size_t)(iw + m) * KK;
    const _Float16* brow = BT + (size_t)j * OUTC * KK;

    for (int kb = 0; kb < KK; kb += 32) {
        FragAB fa;
        const _Float16* pa = arow + kb + hi * 8;
        fa.q[0] = *(const uint4*)(pa);        // K = kb + hi*8 .. +7
        fa.q[1] = *(const uint4*)(pa + 16);   // K = kb + 16 + hi*8 .. +7
        #pragma unroll
        for (int ot = 0; ot < 8; ++ot) {
            FragAB fb;
            const _Float16* pb = brow + (size_t)(ot * 16 + m) * KK + kb + hi * 8;
            fb.q[0] = *(const uint4*)(pb);       // K = kb + hi*8 .. +7
            fb.q[1] = *(const uint4*)(pb + 16);  // K = kb + 16 + hi*8 .. +7
            acc[ot] = __builtin_amdgcn_wmma_f32_16x16x32_f16(
                false, fa.h, false, fb.h, (short)0, acc[ot], false, false);
        }
    }

    #pragma unroll
    for (int ot = 0; ot < 8; ++ot) {
        const int o  = ot * 16 + m;
        const float bo = bias[o];
        #pragma unroll
        for (int r = 0; r < 8; ++r) {
            const int ii  = iw + hi * 8 + r;
            const int row = swapped ? (rowBase + j)  : (rowBase + ii);
            const int col = swapped ? (colBase + ii) : (colBase + j);
            __builtin_nontemporal_store(acc[ot][r] + bo,
                out + ((size_t)row * NTOT + col) * OUTC + o);
        }
    }
}

// ---------------------------------------------------------------------------
extern "C" void kernel_launch(void* const* d_in, const int* in_sizes, int n_in,
                              void* d_out, int out_size, void* d_ws, size_t ws_size,
                              hipStream_t stream) {
    const float* rec     = (const float*)d_in[0];
    const float* lig     = (const float*)d_in[1];
    // d_in[2] (pw_rep) only supplies the shape; never read.
    const float* r_g     = (const float*)d_in[3];
    const float* r_b     = (const float*)d_in[4];
    const float* l_g     = (const float*)d_in[5];
    const float* l_b     = (const float*)d_in[6];
    const float* W_r     = (const float*)d_in[7];
    const float* b_r     = (const float*)d_in[8];
    const float* W_l     = (const float*)d_in[9];
    const float* b_l     = (const float*)d_in[10];
    const float* W_rr    = (const float*)d_in[11];
    const float* b_rr    = (const float*)d_in[12];
    const float* W_rl    = (const float*)d_in[13];
    const float* b_rl    = (const float*)d_in[14];
    const float* W_lr    = (const float*)d_in[15];
    const float* b_lr    = (const float*)d_in[16];
    const float* W_ll    = (const float*)d_in[17];
    const float* b_ll    = (const float*)d_in[18];
    float* out = (float*)d_out;

    // Workspace carve-up (all offsets 256B-aligned).
    char* ws = (char*)d_ws;
    size_t off = 0;
    _Float16* a_rr   = (_Float16*)(ws + off); off += (size_t)NRR * MIDC * 2;          // r_ri  f16
    _Float16* a_rl   = (_Float16*)(ws + off); off += (size_t)NRR * MIDC * 2;          // r_li  f16
    _Float16* a_lr   = (_Float16*)(ws + off); off += (size_t)NRR * MIDC * 2;          // r_lj  f16
    _Float16* a_ll   = (_Float16*)(ws + off); off += (size_t)NLL * MM * MIDC * 2;     // l_li  f16 [i][m*32+x]
    float*    r_rj   = (float*)(ws + off);    off += (size_t)NRR * MIDC * 4;
    float*    l_ri_m = (float*)(ws + off);    off += (size_t)NLL * MIDC * 4;
    float*    l_rj_m = (float*)(ws + off);    off += (size_t)NLL * MIDC * 4;
    float*    l_lj   = (float*)(ws + off);    off += (size_t)MM * NLL * MIDC * 4;
    _Float16* bt_rr  = (_Float16*)(ws + off); off += (size_t)NRR * OUTC * MIDC * 2;   // 4 MB
    _Float16* bt_rl  = (_Float16*)(ws + off); off += (size_t)NLL * OUTC * MIDC * 2;   // 2 MB
    _Float16* bt_lr  = (_Float16*)(ws + off); off += (size_t)NLL * OUTC * MIDC * 2;   // 2 MB
    _Float16* bt_ll  = (_Float16*)(ws + off); off += (size_t)NLL * OUTC * MM * MIDC * 2; // 8 MB
    (void)ws_size; (void)in_sizes; (void)n_in; (void)out_size;

    // Phase 1: layernorm + projections
    ln_proj_rec_kernel<<<NRR, 128, 0, stream>>>(rec, r_g, r_b, W_r, b_r,
                                                a_rr, r_rj, a_rl, a_lr);
    ln_proj_lig_kernel<<<NLL, 128, 0, stream>>>(lig, l_g, l_b, W_l, b_l,
                                                l_ri_m, l_rj_m, a_ll, l_lj);

    // Phase 2: per-j weight contractions t[j][o][x] (f16, WMMA-B layout)
    compute_t_kernel<<<NRR, 256, 0, stream>>>(r_rj,   W_rr, bt_rr, MIDC, 0, 1.0f);
    compute_t_kernel<<<NLL, 256, 0, stream>>>(l_rj_m, W_rl, bt_rl, MIDC, 0, 1.0f);
    compute_t_kernel<<<NLL, 256, 0, stream>>>(l_ri_m, W_lr, bt_lr, MIDC, 0, 1.0f);
    for (int m = 0; m < MM; ++m)   // ll: K = m*32+x, fold 1/m into t
        compute_t_kernel<<<NLL, 256, 0, stream>>>(l_lj + (size_t)m * NLL * MIDC,
                                                  W_ll, bt_ll, MM * MIDC, m * MIDC,
                                                  1.0f / MM);

    // Phase 3: pairwise WMMA blocks (store-bound: 302 MB of output)
    // rr: [0:512, 0:512]
    pair_wmma_kernel<<<512 * 4, 256, 0, stream>>>(a_rr, bt_rr, b_rr, out,
                                                  4, NRR, MIDC, 0, 0, 0);
    // rl: [0:512, 512:768]
    pair_wmma_kernel<<<256 * 4, 256, 0, stream>>>(a_rl, bt_rl, b_rl, out,
                                                  4, NLL, MIDC, 0, NRR, 0);
    // lr: [512:768, 0:512] (row from j, col from i)
    pair_wmma_kernel<<<256 * 4, 256, 0, stream>>>(a_lr, bt_lr, b_lr, out,
                                                  4, NLL, MIDC, NRR, 0, 1);
    // ll: [512:768, 512:768], K = 128 (4 WMMA steps)
    pair_wmma_kernel<<<256 * 2, 256, 0, stream>>>(a_ll, bt_ll, b_ll, out,
                                                  2, NLL, MM * MIDC, NRR, NRR, 0);
}
